// AttentionPairBias_34995393528033
// MI455X (gfx1250) — compile-verified, matrix operand order checked by reading, not run on hardware
//
#include <hip/hip_runtime.h>

#define B_   2
#define S_   768
#define CA_  768
#define CS_  384
#define CZ_  128
#define H_   16
#define HD_  48
#define HDP_ 64
#define MR_  1536            // B_*S_
#define SCALE_ 0.14433756729740643f   // 1/sqrt(48)

typedef __bf16 bf16_t;
typedef __attribute__((ext_vector_type(16))) __bf16 v16bf;
typedef __attribute__((ext_vector_type(8)))  __bf16 v8bf;
typedef __attribute__((ext_vector_type(8)))  float  v8f;

static __device__ __forceinline__ float sigm(float x) { return 1.0f / (1.0f + __expf(-x)); }

// Load a 16x32 bf16 WMMA fragment (A operand, or B operand from a pre-transposed
// Bt[n][k] matrix). Per CDNA5 ISA 16-bit A layout: lane&15 = row, lane>>4 picks
// K-halves {k0..k0+7, k0+16..k0+23} / {k0+8..k0+15, k0+24..k0+31}.
static __device__ __forceinline__ v16bf load_frag(const bf16_t* p, int ld, int row0, int k0) {
  int lane = threadIdx.x & 31;
  const bf16_t* q = p + (size_t)(row0 + (lane & 15)) * (size_t)ld + (size_t)(k0 + ((lane >> 4) << 3));
  v8bf lo = *reinterpret_cast<const v8bf*>(q);
  v8bf hi = *reinterpret_cast<const v8bf*>(q + 16);
  return __builtin_shufflevector(lo, hi, 0,1,2,3,4,5,6,7,8,9,10,11,12,13,14,15);
}

static __device__ __forceinline__ void mma(v8f& acc, v16bf a, v16bf b) {
  acc = __builtin_amdgcn_wmma_f32_16x16x32_bf16(false, a, false, b, (short)0, acc, false, false);
}

// ---------------- prep kernels ----------------

// dst[n*K + k] = (bf16) src[k*N + n]   (store W transposed: Bt[n][k])
__global__ void k_tpose(const float* __restrict__ src, bf16_t* __restrict__ dst, int K, int N) {
  int idx = blockIdx.x * 256 + threadIdx.x;
  if (idx < K * N) {
    int n = idx / K, k = idx - n * K;
    dst[idx] = (bf16_t)src[(size_t)k * N + n];
  }
}

__global__ void k_zero(bf16_t* p, int n) {
  int i = blockIdx.x * 256 + threadIdx.x;
  if (i < n) p[i] = (bf16_t)0.0f;
}

// per-row LN of s (384 ch): sn = LN(s)*gamma ; also raw s -> bf16
__global__ void k_prep_s(const float* __restrict__ s, const float* __restrict__ gamma,
                         bf16_t* __restrict__ s_bf, bf16_t* __restrict__ sn_bf) {
  __shared__ float red[128];
  int row = blockIdx.x, t = threadIdx.x;
  const float* sr = s + (size_t)row * CS_;
  float x0 = sr[t], x1 = sr[t + 128], x2 = sr[t + 256];
  red[t] = x0 + x1 + x2; __syncthreads();
  for (int o = 64; o > 0; o >>= 1) { if (t < o) red[t] += red[t + o]; __syncthreads(); }
  float mean = red[0] * (1.0f / CS_); __syncthreads();
  float d0 = x0 - mean, d1 = x1 - mean, d2 = x2 - mean;
  red[t] = d0 * d0 + d1 * d1 + d2 * d2; __syncthreads();
  for (int o = 64; o > 0; o >>= 1) { if (t < o) red[t] += red[t + o]; __syncthreads(); }
  float rinv = rsqrtf(red[0] * (1.0f / CS_) + 1e-5f);
  size_t base = (size_t)row * CS_;
  sn_bf[base + t]       = (bf16_t)(d0 * rinv * gamma[t]);
  sn_bf[base + t + 128] = (bf16_t)(d1 * rinv * gamma[t + 128]);
  sn_bf[base + t + 256] = (bf16_t)(d2 * rinv * gamma[t + 256]);
  s_bf[base + t] = (bf16_t)x0; s_bf[base + t + 128] = (bf16_t)x1; s_bf[base + t + 256] = (bf16_t)x2;
}

// per-row LN of a (768 ch) -> fp32
__global__ void k_prep_a(const float* __restrict__ a, float* __restrict__ lnA) {
  __shared__ float red[256];
  int row = blockIdx.x, t = threadIdx.x;
  const float* ar = a + (size_t)row * CA_;
  float x0 = ar[t], x1 = ar[t + 256], x2 = ar[t + 512];
  red[t] = x0 + x1 + x2; __syncthreads();
  for (int o = 128; o > 0; o >>= 1) { if (t < o) red[t] += red[t + o]; __syncthreads(); }
  float mean = red[0] * (1.0f / CA_); __syncthreads();
  float d0 = x0 - mean, d1 = x1 - mean, d2 = x2 - mean;
  red[t] = d0 * d0 + d1 * d1 + d2 * d2; __syncthreads();
  for (int o = 128; o > 0; o >>= 1) { if (t < o) red[t] += red[t + o]; __syncthreads(); }
  float rinv = rsqrtf(red[0] * (1.0f / CA_) + 1e-5f);
  size_t base = (size_t)row * CA_;
  lnA[base + t] = d0 * rinv; lnA[base + t + 256] = d1 * rinv; lnA[base + t + 512] = d2 * rinv;
}

// ---------------- GEMM kernels (software-pipelined fragment double buffering) ----------------

// a_n = sigmoid(sn@Wg + bg) * LN(a) + sn@Wo    (M=1536,N=768,K=384), wave = 16x32 tile
__global__ void k_ada(const bf16_t* __restrict__ sn, const bf16_t* __restrict__ wgT,
                      const bf16_t* __restrict__ woT, const float* __restrict__ bg,
                      const float* __restrict__ lnA, bf16_t* __restrict__ an) {
  int w = blockIdx.x * 8 + (threadIdx.x >> 5);   // 2304 waves
  int nt = w % 24, mt = w / 24;
  int m0 = mt * 16, n0 = nt * 32;
  v8f accG[2] = {}, accO[2] = {};
  v16bf aC = load_frag(sn, CS_, m0, 0);
  v16bf gC[2], oC[2];
#pragma unroll
  for (int u = 0; u < 2; ++u) {
    gC[u] = load_frag(wgT, CS_, n0 + 16 * u, 0);
    oC[u] = load_frag(woT, CS_, n0 + 16 * u, 0);
  }
  for (int k = 0; k < CS_; k += 32) {
    v16bf aN, gN[2], oN[2];
    if (k + 32 < CS_) {
      aN = load_frag(sn, CS_, m0, k + 32);
#pragma unroll
      for (int u = 0; u < 2; ++u) {
        gN[u] = load_frag(wgT, CS_, n0 + 16 * u, k + 32);
        oN[u] = load_frag(woT, CS_, n0 + 16 * u, k + 32);
      }
    }
#pragma unroll
    for (int u = 0; u < 2; ++u) { mma(accG[u], aC, gC[u]); mma(accO[u], aC, oC[u]); }
    aC = aN;
#pragma unroll
    for (int u = 0; u < 2; ++u) { gC[u] = gN[u]; oC[u] = oN[u]; }
  }
  int lane = threadIdx.x & 31, n = lane & 15, sel = lane >> 4;
#pragma unroll
  for (int u = 0; u < 2; ++u) {
    int col = n0 + 16 * u + n;
#pragma unroll
    for (int r = 0; r < 8; ++r) {
      int row = m0 + r + 8 * sel;
      float gv = sigm(accG[u][r] + bg[col]);
      an[(size_t)row * CA_ + col] = (bf16_t)(gv * lnA[(size_t)row * CA_ + col] + accO[u][r]);
    }
  }
}

// gate = sigmoid(s@Ws_gate + bs_gate)   (M=1536,N=768,K=384), wave = 16x64 tile
__global__ void k_gate(const bf16_t* __restrict__ s_bf, const bf16_t* __restrict__ wsgT,
                       const float* __restrict__ bsg, float* __restrict__ gate) {
  int w = blockIdx.x * 8 + (threadIdx.x >> 5);   // 1152 waves
  int nt = w % 12, mt = w / 12;
  int m0 = mt * 16, n0 = nt * 64;
  v8f acc[4] = {};
  v16bf aC = load_frag(s_bf, CS_, m0, 0);
  v16bf bC[4];
#pragma unroll
  for (int u = 0; u < 4; ++u) bC[u] = load_frag(wsgT, CS_, n0 + 16 * u, 0);
  for (int k = 0; k < CS_; k += 32) {
    v16bf aN, bN[4];
    if (k + 32 < CS_) {
      aN = load_frag(s_bf, CS_, m0, k + 32);
#pragma unroll
      for (int u = 0; u < 4; ++u) bN[u] = load_frag(wsgT, CS_, n0 + 16 * u, k + 32);
    }
#pragma unroll
    for (int u = 0; u < 4; ++u) mma(acc[u], aC, bC[u]);
    aC = aN;
#pragma unroll
    for (int u = 0; u < 4; ++u) bC[u] = bN[u];
  }
  int lane = threadIdx.x & 31, n = lane & 15, sel = lane >> 4;
#pragma unroll
  for (int u = 0; u < 4; ++u) {
    int col = n0 + 16 * u + n;
#pragma unroll
    for (int r = 0; r < 8; ++r)
      gate[(size_t)(m0 + r + 8 * sel) * CA_ + col] = sigm(acc[u][r] + bsg[col]);
  }
}

// fused q/k/v/g projections (M=1536,N=3072,K=768), wave = 16x64 tile
__global__ void k_qkvg(const bf16_t* __restrict__ an, const bf16_t* __restrict__ wqT,
                       const bf16_t* __restrict__ wkT, const bf16_t* __restrict__ wvT,
                       const bf16_t* __restrict__ wgT, const float* __restrict__ bq,
                       bf16_t* __restrict__ qpad, bf16_t* __restrict__ kpad,
                       bf16_t* __restrict__ vTr, bf16_t* __restrict__ gmat) {
  int w = blockIdx.x * 8 + (threadIdx.x >> 5);   // 4608 waves
  int nt = w % 48, mt = w / 48;
  int m0 = mt * 16, n0g = nt * 64;
  int which = n0g / CA_, n0 = n0g % CA_;         // 64 | 768 -> no straddle
  const bf16_t* wt = (which == 0) ? wqT : (which == 1) ? wkT : (which == 2) ? wvT : wgT;
  v8f acc[4] = {};
  v16bf aC = load_frag(an, CA_, m0, 0);
  v16bf bC[4];
#pragma unroll
  for (int u = 0; u < 4; ++u) bC[u] = load_frag(wt, CA_, n0 + 16 * u, 0);
  for (int k = 0; k < CA_; k += 32) {
    v16bf aN, bN[4];
    if (k + 32 < CA_) {
      aN = load_frag(an, CA_, m0, k + 32);
#pragma unroll
      for (int u = 0; u < 4; ++u) bN[u] = load_frag(wt, CA_, n0 + 16 * u, k + 32);
    }
#pragma unroll
    for (int u = 0; u < 4; ++u) mma(acc[u], aC, bC[u]);
    aC = aN;
#pragma unroll
    for (int u = 0; u < 4; ++u) bC[u] = bN[u];
  }
  int lane = threadIdx.x & 31, n = lane & 15, sel = lane >> 4;
#pragma unroll
  for (int u = 0; u < 4; ++u) {
    int c = n0 + 16 * u + n, h = c / HD_, d = c - h * HD_;
#pragma unroll
    for (int r = 0; r < 8; ++r) {
      int row = m0 + r + 8 * sel;
      int b = row / S_, i = row - b * S_;
      if (which == 0)      qpad[((size_t)(b * H_ + h) * S_ + i) * HDP_ + d] = (bf16_t)((acc[u][r] + bq[c]) * SCALE_);
      else if (which == 1) kpad[((size_t)(b * H_ + h) * S_ + i) * HDP_ + d] = (bf16_t)acc[u][r];
      else if (which == 2) vTr[((size_t)(b * H_ + h) * HD_ + d) * S_ + i]   = (bf16_t)acc[u][r];
      else                 gmat[(size_t)row * CA_ + c] = (bf16_t)sigm(acc[u][r]);
    }
  }
}

// pair bias: logits[b,i,h,j] = LN(z[b,j,i,:])*gz+bz @ Wb  + beta[b,j,i,h]
// Async-DMA the 64KB z tile into LDS (ASYNCcnt), LN out of LDS, WMMA from LDS bf16.
__global__ void k_bias(const float* __restrict__ z, const float* __restrict__ beta,
                       const float* __restrict__ gz, const float* __restrict__ bz,
                       const bf16_t* __restrict__ wbT, float* __restrict__ L) {
  __shared__ __attribute__((aligned(16))) char smem[128 * CZ_ * 4 + 128 * CZ_ * 2];
  float*  zt = (float*)smem;                       // LDS byte offset 0
  bf16_t* nz = (bf16_t*)(smem + 128 * CZ_ * 4);
  int idx = blockIdx.x;
  int jblk = idx % 6, i = (idx / 6) % S_, b = idx / (6 * S_);
  int t = threadIdx.x;                             // 128 threads
  {
    int j = jblk * 128 + t;
    const char* gbase = (const char*)(z + (((size_t)b * S_ + j) * S_ + i) * CZ_);
    unsigned lds0 = (unsigned)t * (CZ_ * 4u);      // this thread's row in zt
#pragma unroll 4
    for (int c = 0; c < 32; ++c) {
      unsigned lo = lds0 + (unsigned)c * 16u;
      unsigned long long ga = (unsigned long long)(uintptr_t)(gbase + c * 16);
      asm volatile("global_load_async_to_lds_b128 %0, %1, off"
                   :: "v"(lo), "v"(ga) : "memory");
    }
    asm volatile("s_wait_asynccnt 0" ::: "memory");
  }
  __syncthreads();
  {
    const float4* zp = (const float4*)(zt + (size_t)t * CZ_);
    float sum = 0.f, sq = 0.f;
#pragma unroll 8
    for (int c = 0; c < 32; ++c) {
      float4 v4 = zp[c];
      sum += v4.x + v4.y + v4.z + v4.w;
      sq  += v4.x * v4.x + v4.y * v4.y + v4.z * v4.z + v4.w * v4.w;
    }
    float mean = sum * (1.0f / CZ_);
    float var  = fmaxf(sq * (1.0f / CZ_) - mean * mean, 0.0f);
    float rinv = rsqrtf(var + 1e-5f);
#pragma unroll 8
    for (int c = 0; c < 32; ++c) {
      float4 v4 = zp[c]; int c0 = c * 4;
      nz[t * CZ_ + c0 + 0] = (bf16_t)((v4.x - mean) * rinv * gz[c0 + 0] + bz[c0 + 0]);
      nz[t * CZ_ + c0 + 1] = (bf16_t)((v4.y - mean) * rinv * gz[c0 + 1] + bz[c0 + 1]);
      nz[t * CZ_ + c0 + 2] = (bf16_t)((v4.z - mean) * rinv * gz[c0 + 2] + bz[c0 + 2]);
      nz[t * CZ_ + c0 + 3] = (bf16_t)((v4.w - mean) * rinv * gz[c0 + 3] + bz[c0 + 3]);
    }
  }
  __syncthreads();
  int wv = t >> 5, lane = t & 31, n = lane & 15, sel = lane >> 4;
  for (int qi = 0; qi < 2; ++qi) {
    int j0 = (wv * 2 + qi) * 16;
    int jg = jblk * 128 + j0 + n;
    v8f acc;
#pragma unroll
    for (int r = 0; r < 8; ++r) {
      int h = r + 8 * sel;
      acc[r] = beta[(((size_t)b * S_ + jg) * S_ + i) * H_ + h];   // C = beta (transposed view)
    }
    for (int k = 0; k < CZ_; k += 32)
      mma(acc, load_frag(wbT, CZ_, 0, k), load_frag(nz, CZ_, j0, k));
#pragma unroll
    for (int r = 0; r < 8; ++r) {
      int h = r + 8 * sel;
      L[(((size_t)b * S_ + i) * H_ + h) * S_ + jg] = acc[r];
    }
  }
}

// logits += q.kT (K=64, zero padded); wave does 16x64 j-tile, all fragments preloaded
__global__ void k_qk(const bf16_t* __restrict__ qpad, const bf16_t* __restrict__ kpad,
                     float* __restrict__ L) {
  int w = blockIdx.x * 8 + (threadIdx.x >> 5);   // 18432 waves
  int jt = w % 12, it = (w / 12) % 48, h = (w / 576) % 16, b = w / 9216;
  const bf16_t* qb = qpad + (size_t)(b * H_ + h) * S_ * HDP_;
  const bf16_t* kb = kpad + (size_t)(b * H_ + h) * S_ * HDP_;
  float* Lb = L + (size_t)b * S_ * H_ * S_;
  int i0 = it * 16, j0 = jt * 64;
  int lane = threadIdx.x & 31, n = lane & 15, sel = lane >> 4;
  v16bf q0 = load_frag(qb, HDP_, i0, 0);
  v16bf q1 = load_frag(qb, HDP_, i0, 32);
  v16bf kf0[4], kf1[4];
#pragma unroll
  for (int u = 0; u < 4; ++u) {
    kf0[u] = load_frag(kb, HDP_, j0 + 16 * u, 0);
    kf1[u] = load_frag(kb, HDP_, j0 + 16 * u, 32);
  }
  v8f acc[4];
#pragma unroll
  for (int u = 0; u < 4; ++u)
#pragma unroll
    for (int r = 0; r < 8; ++r)
      acc[u][r] = Lb[((size_t)(i0 + r + 8 * sel) * H_ + h) * S_ + j0 + 16 * u + n];
#pragma unroll
  for (int u = 0; u < 4; ++u) {
    mma(acc[u], q0, kf0[u]);
    mma(acc[u], q1, kf1[u]);
  }
#pragma unroll
  for (int u = 0; u < 4; ++u)
#pragma unroll
    for (int r = 0; r < 8; ++r)
      Lb[((size_t)(i0 + r + 8 * sel) * H_ + h) * S_ + j0 + 16 * u + n] = acc[u][r];
}

// softmax over j; write attn bf16 in-place over the fp32 row
__global__ void k_softmax(float* __restrict__ L) {
  __shared__ float red[256];
  float* Lr = L + (size_t)blockIdx.x * S_;
  int t = threadIdx.x;
  float x0 = Lr[t], x1 = Lr[t + 256], x2 = Lr[t + 512];
  red[t] = fmaxf(fmaxf(x0, x1), x2); __syncthreads();
  for (int o = 128; o > 0; o >>= 1) { if (t < o) red[t] = fmaxf(red[t], red[t + o]); __syncthreads(); }
  float m = red[0]; __syncthreads();
  float e0 = __expf(x0 - m), e1 = __expf(x1 - m), e2 = __expf(x2 - m);
  red[t] = e0 + e1 + e2; __syncthreads();
  for (int o = 128; o > 0; o >>= 1) { if (t < o) red[t] += red[t + o]; __syncthreads(); }
  float inv = 1.0f / red[0]; __syncthreads();
  bf16_t* Ar = (bf16_t*)Lr;
  Ar[t] = (bf16_t)(e0 * inv); Ar[t + 256] = (bf16_t)(e1 * inv); Ar[t + 512] = (bf16_t)(e2 * inv);
}

// o = attn @ v ; go = g*o ; wave covers all 48 d-cols (3 tiles), pipelined
__global__ void k_av(const float* __restrict__ L, const bf16_t* __restrict__ vTr,
                     const bf16_t* __restrict__ gmat, bf16_t* __restrict__ go) {
  int w = blockIdx.x * 8 + (threadIdx.x >> 5);   // 1536 waves
  int it = w % 48, h = (w / 48) % 16, b = w / 768;
  const bf16_t* A  = (const bf16_t*)L + 2 * ((size_t)b * S_ * H_ * S_ + (size_t)h * S_);
  const int lda = 2 * H_ * S_;                   // bf16 rows embedded in fp32 rows
  const bf16_t* Bt = vTr + (size_t)(b * H_ + h) * HD_ * S_;
  int i0 = it * 16;
  v8f acc[3] = {};
  v16bf aC = load_frag(A, lda, i0, 0);
  v16bf bC[3];
#pragma unroll
  for (int u = 0; u < 3; ++u) bC[u] = load_frag(Bt, S_, 16 * u, 0);
  for (int k = 0; k < S_; k += 32) {
    v16bf aN, bN[3];
    if (k + 32 < S_) {
      aN = load_frag(A, lda, i0, k + 32);
#pragma unroll
      for (int u = 0; u < 3; ++u) bN[u] = load_frag(Bt, S_, 16 * u, k + 32);
    }
#pragma unroll
    for (int u = 0; u < 3; ++u) mma(acc[u], aC, bC[u]);
    aC = aN;
#pragma unroll
    for (int u = 0; u < 3; ++u) bC[u] = bN[u];
  }
  int lane = threadIdx.x & 31, n = lane & 15, sel = lane >> 4;
#pragma unroll
  for (int u = 0; u < 3; ++u) {
    int c = h * HD_ + 16 * u + n;
#pragma unroll
    for (int r = 0; r < 8; ++r) {
      size_t row = (size_t)b * S_ + i0 + r + 8 * sel;
      go[row * CA_ + c] = (bf16_t)((float)gmat[row * CA_ + c] * acc[u][r]);
    }
  }
}

// out = gate * (go @ Wout)   (M=1536,N=768,K=768), wave = 16x64 tile
__global__ void k_out(const bf16_t* __restrict__ go, const bf16_t* __restrict__ woutT,
                      const float* __restrict__ gate, float* __restrict__ out) {
  int w = blockIdx.x * 8 + (threadIdx.x >> 5);   // 1152 waves
  int nt = w % 12, mt = w / 12;
  int m0 = mt * 16, n0 = nt * 64;
  v8f acc[4] = {};
  v16bf aC = load_frag(go, CA_, m0, 0);
  v16bf bC[4];
#pragma unroll
  for (int u = 0; u < 4; ++u) bC[u] = load_frag(woutT, CA_, n0 + 16 * u, 0);
  for (int k = 0; k < CA_; k += 32) {
    v16bf aN, bN[4];
    if (k + 32 < CA_) {
      aN = load_frag(go, CA_, m0, k + 32);
#pragma unroll
      for (int u = 0; u < 4; ++u) bN[u] = load_frag(woutT, CA_, n0 + 16 * u, k + 32);
    }
#pragma unroll
    for (int u = 0; u < 4; ++u) mma(acc[u], aC, bC[u]);
    aC = aN;
#pragma unroll
    for (int u = 0; u < 4; ++u) bC[u] = bN[u];
  }
  int lane = threadIdx.x & 31, n = lane & 15, sel = lane >> 4;
#pragma unroll
  for (int u = 0; u < 4; ++u) {
    int col = n0 + 16 * u + n;
#pragma unroll
    for (int r = 0; r < 8; ++r) {
      size_t row = (size_t)(m0 + r + 8 * sel);
      out[row * CA_ + col] = gate[row * CA_ + col] * acc[u][r];
    }
  }
}

// ---------------- launcher ----------------

extern "C" void kernel_launch(void* const* d_in, const int* in_sizes, int n_in,
                              void* d_out, int out_size, void* d_ws, size_t ws_size,
                              hipStream_t stream) {
  const float* a       = (const float*)d_in[0];
  const float* s       = (const float*)d_in[1];
  const float* z       = (const float*)d_in[2];
  const float* beta    = (const float*)d_in[3];
  const float* gamma_s = (const float*)d_in[4];
  const float* W_ada_g = (const float*)d_in[5];
  const float* b_ada_g = (const float*)d_in[6];
  const float* W_ada_o = (const float*)d_in[7];
  const float* Wq      = (const float*)d_in[8];
  const float* bq      = (const float*)d_in[9];
  const float* Wk      = (const float*)d_in[10];
  const float* Wv      = (const float*)d_in[11];
  const float* Wb      = (const float*)d_in[12];
  const float* gz      = (const float*)d_in[13];
  const float* bz      = (const float*)d_in[14];
  const float* Wg      = (const float*)d_in[15];
  const float* Wout    = (const float*)d_in[16];
  const float* Ws_gate = (const float*)d_in[17];
  const float* bs_gate = (const float*)d_in[18];
  float* out = (float*)d_out;

  char* base = (char*)d_ws; size_t off = 0;
  auto carve = [&](size_t bytes) -> void* {
    void* p = base + off; off += (bytes + 255) & ~(size_t)255; return p;
  };
  bf16_t* wadaGT = (bf16_t*)carve((size_t)CA_ * CS_ * 2);
  bf16_t* wadaOT = (bf16_t*)carve((size_t)CA_ * CS_ * 2);
  bf16_t* wsgT   = (bf16_t*)carve((size_t)CA_ * CS_ * 2);
  bf16_t* wqT    = (bf16_t*)carve((size_t)CA_ * CA_ * 2);
  bf16_t* wkT    = (bf16_t*)carve((size_t)CA_ * CA_ * 2);
  bf16_t* wvTw   = (bf16_t*)carve((size_t)CA_ * CA_ * 2);
  bf16_t* wgT    = (bf16_t*)carve((size_t)CA_ * CA_ * 2);
  bf16_t* woutT  = (bf16_t*)carve((size_t)CA_ * CA_ * 2);
  bf16_t* wbT    = (bf16_t*)carve((size_t)H_ * CZ_ * 2);
  bf16_t* s_bf   = (bf16_t*)carve((size_t)MR_ * CS_ * 2);
  bf16_t* sn_bf  = (bf16_t*)carve((size_t)MR_ * CS_ * 2);
  float*  lnA    = (float*) carve((size_t)MR_ * CA_ * 4);
  bf16_t* an_bf  = (bf16_t*)carve((size_t)MR_ * CA_ * 2);
  float*  gate   = (float*) carve((size_t)MR_ * CA_ * 4);
  bf16_t* qpad   = (bf16_t*)carve((size_t)B_ * H_ * S_ * HDP_ * 2);
  bf16_t* kpad   = (bf16_t*)carve((size_t)B_ * H_ * S_ * HDP_ * 2);
  bf16_t* vTr    = (bf16_t*)carve((size_t)B_ * H_ * HD_ * S_ * 2);
  bf16_t* gmat   = (bf16_t*)carve((size_t)MR_ * CA_ * 2);
  bf16_t* go     = (bf16_t*)carve((size_t)MR_ * CA_ * 2);
  float*  L      = (float*) carve((size_t)B_ * S_ * H_ * S_ * 4);

  auto tp = [&](const float* src, bf16_t* dst, int K, int N) {
    k_tpose<<<(K * N + 255) / 256, 256, 0, stream>>>(src, dst, K, N);
  };
  tp(W_ada_g, wadaGT, CS_, CA_);
  tp(W_ada_o, wadaOT, CS_, CA_);
  tp(Ws_gate, wsgT,   CS_, CA_);
  tp(Wq, wqT, CA_, CA_);  tp(Wk, wkT, CA_, CA_);
  tp(Wv, wvTw, CA_, CA_); tp(Wg, wgT, CA_, CA_);
  tp(Wout, woutT, CA_, CA_);
  tp(Wb, wbT, CZ_, H_);

  int npad = B_ * H_ * S_ * HDP_ * 2;                       // qpad+kpad contiguous
  k_zero<<<(npad + 255) / 256, 256, 0, stream>>>(qpad, npad);

  k_prep_s<<<MR_, 128, 0, stream>>>(s, gamma_s, s_bf, sn_bf);
  k_prep_a<<<MR_, 256, 0, stream>>>(a, lnA);
  k_ada <<<288, 256, 0, stream>>>(sn_bf, wadaGT, wadaOT, b_ada_g, lnA, an_bf);
  k_gate<<<144, 256, 0, stream>>>(s_bf, wsgT, bs_gate, gate);
  k_qkvg<<<576, 256, 0, stream>>>(an_bf, wqT, wkT, wvTw, wgT, bq, qpad, kpad, vTr, gmat);
  k_bias<<<B_ * S_ * 6, 128, 0, stream>>>(z, beta, gz, bz, wbT, L);
  k_qk  <<<2304, 256, 0, stream>>>(qpad, kpad, L);
  k_softmax<<<B_ * S_ * H_, 256, 0, stream>>>(L);
  k_av  <<<192, 256, 0, stream>>>(L, vTr, gmat, go);
  k_out <<<144, 256, 0, stream>>>(go, woutT, gate, out);
  (void)in_sizes; (void)n_in; (void)out_size; (void)ws_size;
}
